// CoxSurvLoss_81423989997611
// MI455X (gfx1250) — matrix operand order repaired and checked by reading, compiled, not run on hardware
//
#include <hip/hip_runtime.h>

// CDNA5 / gfx1250, wave32. Cox partial-likelihood loss, N = 8192.
//   risk[i] = sum_j (time[j] >= time[i]) * exp(theta[j])   -> bf16 WMMA matvec
//   loss    = -(1/N) * sum_i c_i * (theta_i - log(risk[i]))

typedef __attribute__((ext_vector_type(16))) __bf16 v16bf;
typedef __attribute__((ext_vector_type(8)))  float  v8f;

constexpr int kN        = 8192;   // problem size (hazards/time/c length)
constexpr int kWavesPB  = 8;      // 8 waves * 32 lanes = 256 threads/block
constexpr int kTileM    = 16;     // i-rows per wave tile
constexpr int kTileK    = 32;     // j-chunk per WMMA (16x16x32 bf16)

// ---------------------------------------------------------------- kernel 1 --
// exp(theta) -> bf16 bits in workspace
__global__ void cox_exp_kernel(const float* __restrict__ theta,
                               unsigned short* __restrict__ exp_bits, int n) {
    int i = blockIdx.x * blockDim.x + threadIdx.x;
    if (i < n) {
        __bf16 v = (__bf16)__expf(theta[i]);
        exp_bits[i] = __builtin_bit_cast(unsigned short, v);
    }
}

// ---------------------------------------------------------------- kernel 2 --
// risk[i] = sum_j mask(i,j) * exp(theta[j]) via v_wmma_f32_16x16x32_bf16.
// A (16x32): exp(theta[j..j+31]) replicated over all 16 M-rows.
// B (32x16): mask tile, B[k,n] = (time[j+k] >= time[i_base+n]).
// D[m,n] = partial risk for i_base+n (identical over m); read row M=0.
__global__ void __launch_bounds__(kWavesPB * 32)
cox_risk_wmma_kernel(const float* __restrict__ time_g,
                     const unsigned short* __restrict__ exp_bits,
                     float* __restrict__ risk) {
    __shared__ float  s_time[kN];   // 32 KB
    __shared__ __bf16 s_exp[kN];    // 16 KB   (48 KB total; WGP has 320 KB)

    const int tid = threadIdx.x;

    // Seed the staging loads (gfx1250 global_prefetch_b8 path).
    __builtin_prefetch(time_g + tid * 64, 0, 1);

    // Stage the entire time[] and exp(theta)[] arrays into LDS once.
    for (int i = tid; i < kN / 4; i += kWavesPB * 32) {
        ((float4*)s_time)[i] = ((const float4*)time_g)[i];
    }
    for (int i = tid; i < kN / 8; i += kWavesPB * 32) {
        ((uint4*)s_exp)[i] = ((const uint4*)exp_bits)[i];
    }
    __syncthreads();

    const int lane   = tid & 31;          // wave32
    const int wave   = tid >> 5;
    const int tile   = blockIdx.x * kWavesPB + wave;   // 0..511
    const int i_base = tile * kTileM;

    const int n = lane & 15;              // B/D column index for this lane
    const int g = lane >> 4;              // lane half-group (K striping)
    const float t_i = s_time[i_base + n];

    const __bf16 one  = (__bf16)1.0f;
    const __bf16 zero = (__bf16)0.0f;

    v8f acc = {};
    for (int j = 0; j < kN; j += kTileK) {
        v16bf a, b;
#pragma unroll
        for (int v = 0; v < 8; ++v) {
            // A fragment: 16-bit A 16x32 layout. Lane holds M=lane%16 (value
            // is M-independent here). VGPR v, group g -> K base:
            //   K = 16*(v>>2) + 8*g + 2*(v&3)
            const int kA = ((v >> 2) << 4) + (g << 3) + ((v & 3) << 1);
            a[2 * v]     = s_exp[j + kA];
            a[2 * v + 1] = s_exp[j + kA + 1];

            // B fragment: same K order as A;
            // VGPR v half e -> K = 16*(v>>2) + 8*g + 2*(v&3) + e. Lane column = n.
            const int kB = ((v >> 2) << 4) + (g << 3) + ((v & 3) << 1);
            b[2 * v]     = (s_time[j + kB]     >= t_i) ? one : zero;
            b[2 * v + 1] = (s_time[j + kB + 1] >= t_i) ? one : zero;
        }
        // (neg_a, A, neg_b, B, c_mod, C, reuse_a, reuse_b)
        acc = __builtin_amdgcn_wmma_f32_16x16x32_bf16(
            false, a, false, b, (short)0, acc, false, false);
    }

    // D row M=0 lives in VGPR0 of lanes 0-15 (N = lane).
    if (lane < 16) {
        risk[i_base + lane] = acc[0];
    }
}

// ---------------------------------------------------------------- kernel 3 --
// loss = -(1/N) * sum_i c_i * (theta_i - log(risk_i))
__global__ void cox_loss_reduce_kernel(const float* __restrict__ theta,
                                       const int* __restrict__ c,
                                       const float* __restrict__ risk,
                                       float* __restrict__ out, int n) {
    __shared__ float red[256];
    float s = 0.0f;
    for (int i = threadIdx.x; i < n; i += blockDim.x) {
        if (c[i] != 0) {
            s += theta[i] - __logf(risk[i]);
        }
    }
    red[threadIdx.x] = s;
    __syncthreads();
    for (int stride = 128; stride > 0; stride >>= 1) {
        if (threadIdx.x < stride) red[threadIdx.x] += red[threadIdx.x + stride];
        __syncthreads();
    }
    if (threadIdx.x == 0) {
        out[0] = -red[0] / (float)n;
    }
}

// ------------------------------------------------------------------- launch --
extern "C" void kernel_launch(void* const* d_in, const int* in_sizes, int n_in,
                              void* d_out, int out_size, void* d_ws, size_t ws_size,
                              hipStream_t stream) {
    const float* hazards = (const float*)d_in[0];  // [N,1] flat -> theta
    const float* time_g  = (const float*)d_in[1];  // [N]
    const int*   c       = (const int*)d_in[2];    // [N]
    float*       out     = (float*)d_out;          // scalar f32 loss

    // Workspace: [0, 16KB) exp(theta) bf16 bits; [16KB, 48KB) risk f32.
    unsigned short* exp_bits = (unsigned short*)d_ws;
    float*          risk     = (float*)((char*)d_ws + kN * sizeof(unsigned short));

    const int n = kN;  // reference fixes N = 8192

    cox_exp_kernel<<<n / 256, 256, 0, stream>>>(hazards, exp_bits, n);

    // 512 row-tiles of 16, 8 waves (tiles) per 256-thread block -> 64 blocks.
    cox_risk_wmma_kernel<<<(n / kTileM) / kWavesPB, kWavesPB * 32, 0, stream>>>(
        time_g, exp_bits, risk);

    cox_loss_reduce_kernel<<<1, 256, 0, stream>>>(hazards, c, risk, out, n);
}
